// RNNUsingSimpleRNN_35811437314456
// MI455X (gfx1250) — matrix operand order repaired and checked by reading, compiled, not run on hardware
//
#include <hip/hip_runtime.h>
#include <hip/hip_bf16.h>
#include <math.h>

#define BATCH   2048
#define SEQ     80
#define EMB_D   100
#define EMB_PAD 128
#define UNITS   512

typedef _Float16 v8h  __attribute__((ext_vector_type(8)));
typedef _Float16 v16h __attribute__((ext_vector_type(16)));
typedef float    v8f  __attribute__((ext_vector_type(8)));

// ---------------------------------------------------------------------------
// WMMA fragment loaders (wave32, v_wmma_f32_16x16x32_f16 layouts)
// A: 16x32 (MxK) f16. lane L holds row M=L%16; VGPR pairs hold K:
//    lanes 0-15: K=[0..7] then [16..23]; lanes 16-31: K=[8..15] then [24..31]
// ---------------------------------------------------------------------------
__device__ __forceinline__ v16h load_a_frag(const _Float16* __restrict__ A,
                                            int lda, int row_base, int k0, int lane) {
  const int row = row_base + (lane & 15);
  const int kb  = k0 + ((lane >> 4) << 3);
  const _Float16* p = A + (size_t)row * lda + kb;
  v8h lo = *(const v8h*)p;          // 16B load
  v8h hi = *(const v8h*)(p + 16);   // 16B load
  return __builtin_shufflevector(lo, hi, 0,1,2,3,4,5,6,7,8,9,10,11,12,13,14,15);
}

// B: 32x16 (KxN) f16. BT stored N-major over K (ldb = K stride).
// lane L holds col N=L%16; lanes 0-15: K=[0..7] then [16..23]; lanes 16-31: K=[8..15] then [24..31]
__device__ __forceinline__ v16h load_b_frag(const _Float16* __restrict__ BT,
                                            int ldb, int n_base, int k0, int lane) {
  const int n  = n_base + (lane & 15);
  const int kb = k0 + ((lane >> 4) << 3);
  const _Float16* p = BT + (size_t)n * ldb + kb;
  v8h lo = *(const v8h*)p;
  v8h hi = *(const v8h*)(p + 16);
  return __builtin_shufflevector(lo, hi, 0,1,2,3,4,5,6,7,8,9,10,11,12,13,14,15);
}

#define WMMA_F16(a, b, c) \
  __builtin_amdgcn_wmma_f32_16x16x32_f16(false, (a), false, (b), (short)0, (c), false, false)

// ---------------------------------------------------------------------------
// One RNN timestep for one layer: Hnew = tanh(X @ W + Hold @ U + bias)
// X:[2048,k_in] f16 row-major, WT:[512,k_in] f16 (N-major), Hold:[2048,512] f16,
// UT:[512,512] f16 (N-major), bias:[512] f32, Hnew:[2048,512] f16.
// Block = 8 waves (256 thr); wave tile 32x32; block tile 128(M) x 64(N).
// Grid = (2048/128, 512/64) = (16, 8).
// ---------------------------------------------------------------------------
__global__ __launch_bounds__(256) void rnn_step_kernel(
    const _Float16* __restrict__ X, int k_in,
    const _Float16* __restrict__ WT,
    const _Float16* __restrict__ Hold,
    const _Float16* __restrict__ UT,
    const float*    __restrict__ bias,
    _Float16*       __restrict__ Hnew) {
  const int lane  = threadIdx.x & 31;
  const int wave  = threadIdx.x >> 5;
  const int waveM = wave & 3;
  const int waveN = wave >> 2;
  const int m0 = blockIdx.x * 128 + waveM * 32;
  const int n0 = blockIdx.y * 64  + waveN * 32;

  v8f acc[2][2] = {};

  // Phase 1: X @ W  (K = k_in: 128 padded for layer 1, 512 otherwise)
  for (int k0 = 0; k0 < k_in; k0 += 32) {
    v16h a0 = load_a_frag(X, k_in, m0,      k0, lane);
    v16h a1 = load_a_frag(X, k_in, m0 + 16, k0, lane);
    v16h b0 = load_b_frag(WT, k_in, n0,      k0, lane);
    v16h b1 = load_b_frag(WT, k_in, n0 + 16, k0, lane);
    acc[0][0] = WMMA_F16(a0, b0, acc[0][0]);
    acc[0][1] = WMMA_F16(a0, b1, acc[0][1]);
    acc[1][0] = WMMA_F16(a1, b0, acc[1][0]);
    acc[1][1] = WMMA_F16(a1, b1, acc[1][1]);
  }
  // Phase 2: Hold @ U  (K = 512)
  for (int k0 = 0; k0 < UNITS; k0 += 32) {
    v16h a0 = load_a_frag(Hold, UNITS, m0,      k0, lane);
    v16h a1 = load_a_frag(Hold, UNITS, m0 + 16, k0, lane);
    v16h b0 = load_b_frag(UT, UNITS, n0,      k0, lane);
    v16h b1 = load_b_frag(UT, UNITS, n0 + 16, k0, lane);
    acc[0][0] = WMMA_F16(a0, b0, acc[0][0]);
    acc[0][1] = WMMA_F16(a0, b1, acc[0][1]);
    acc[1][0] = WMMA_F16(a1, b0, acc[1][0]);
    acc[1][1] = WMMA_F16(a1, b1, acc[1][1]);
  }

  // Epilogue: +bias, tanh, store f16.
  // C layout: lane holds col n = L%16; VGPR v holds row m = v + (L/16)*8.
  const int rowoff = (lane >> 4) << 3;
  const int coloff = lane & 15;
  #pragma unroll
  for (int mi = 0; mi < 2; ++mi) {
    #pragma unroll
    for (int ni = 0; ni < 2; ++ni) {
      const int n  = n0 + ni * 16 + coloff;
      const float bn = bias[n];
      #pragma unroll
      for (int v = 0; v < 8; ++v) {
        const int m = m0 + mi * 16 + rowoff + v;
        const float val = tanhf(acc[mi][ni][v] + bn);
        Hnew[(size_t)m * UNITS + n] = (_Float16)val;
      }
    }
  }
}

// ---------------------------------------------------------------------------
// Embedding gather -> f16, time-major [T][B][EMB_PAD], cols >= 100 zeroed.
// ---------------------------------------------------------------------------
__global__ __launch_bounds__(256) void embed_kernel(const int* __restrict__ tokens,
                                                    const float* __restrict__ emb,
                                                    _Float16* __restrict__ X) {
  const size_t idx = (size_t)blockIdx.x * 256 + threadIdx.x; // T*B*128
  if (idx >= (size_t)SEQ * BATCH * EMB_PAD) return;
  const int k = (int)(idx & (EMB_PAD - 1));
  const int b = (int)((idx >> 7) & (BATCH - 1));
  const int t = (int)(idx >> 18); // 7 + 11
  float v = 0.0f;
  if (k < EMB_D) {
    const int tok = tokens[b * SEQ + t];
    v = emb[(size_t)tok * EMB_D + k];
  }
  X[idx] = (_Float16)v;
}

// Transpose + convert: WT[n*ldk + k] = (f16) W[k*N + n], zero-pad k in [K, ldk)
__global__ __launch_bounds__(256) void transpose_convert_kernel(
    const float* __restrict__ W, _Float16* __restrict__ WT, int K, int N, int ldk) {
  const int idx = blockIdx.x * 256 + threadIdx.x;
  if (idx >= N * ldk) return;
  const int k = idx % ldk;
  const int n = idx / ldk;
  WT[idx] = (k < K) ? (_Float16)W[(size_t)k * N + n] : (_Float16)0.0f;
}

__global__ __launch_bounds__(256) void zero_f16_kernel(_Float16* __restrict__ p, int count) {
  const int idx = blockIdx.x * 256 + threadIdx.x;
  if (idx < count) p[idx] = (_Float16)0.0f;
}

// out[b] = sigmoid(dot(H[b,:], Wo) + bo)
__global__ __launch_bounds__(256) void output_kernel(const _Float16* __restrict__ H,
                                                     const float* __restrict__ Wo,
                                                     const float* __restrict__ bo,
                                                     float* __restrict__ out) {
  const int b = blockIdx.x * 256 + threadIdx.x;
  if (b >= BATCH) return;
  float s = 0.0f;
  for (int k = 0; k < UNITS; ++k) s += (float)H[(size_t)b * UNITS + k] * Wo[k];
  s += bo[0];
  out[b] = 1.0f / (1.0f + expf(-s));
}

// ---------------------------------------------------------------------------
extern "C" void kernel_launch(void* const* d_in, const int* in_sizes, int n_in,
                              void* d_out, int out_size, void* d_ws, size_t ws_size,
                              hipStream_t stream) {
  const int*   tokens = (const int*)d_in[0];
  const float* emb    = (const float*)d_in[1];
  const float* W[4] = {(const float*)d_in[2], (const float*)d_in[5],
                       (const float*)d_in[8], (const float*)d_in[11]};
  const float* U[4] = {(const float*)d_in[3], (const float*)d_in[6],
                       (const float*)d_in[9], (const float*)d_in[12]};
  const float* bv[4] = {(const float*)d_in[4], (const float*)d_in[7],
                        (const float*)d_in[10], (const float*)d_in[13]};
  const float* Wo = (const float*)d_in[14];
  const float* bo = (const float*)d_in[15];
  float* out = (float*)d_out;

  // Workspace carve-out (256B aligned slices)
  char*  ws  = (char*)d_ws;
  size_t cur = 0;
  auto alloc = [&](size_t bytes) -> char* {
    char* p = ws + cur;
    cur = (cur + bytes + 255) & ~(size_t)255;
    return p;
  };
  _Float16* xbuf = (_Float16*)alloc((size_t)SEQ * BATCH * EMB_PAD * sizeof(_Float16));
  _Float16* wt[4];
  wt[0] = (_Float16*)alloc((size_t)UNITS * EMB_PAD * sizeof(_Float16));
  for (int l = 1; l < 4; ++l) wt[l] = (_Float16*)alloc((size_t)UNITS * UNITS * sizeof(_Float16));
  _Float16* ut[4];
  for (int l = 0; l < 4; ++l) ut[l] = (_Float16*)alloc((size_t)UNITS * UNITS * sizeof(_Float16));
  _Float16* h[4][2];
  for (int l = 0; l < 4; ++l)
    for (int pp = 0; pp < 2; ++pp)
      h[l][pp] = (_Float16*)alloc((size_t)BATCH * UNITS * sizeof(_Float16));

  // --- Weight prep: transpose + f32->f16 (W1 padded K=100 -> 128) ---
  {
    int n = UNITS * EMB_PAD;
    transpose_convert_kernel<<<(n + 255) / 256, 256, 0, stream>>>(W[0], wt[0], EMB_D, UNITS, EMB_PAD);
  }
  for (int l = 1; l < 4; ++l) {
    int n = UNITS * UNITS;
    transpose_convert_kernel<<<(n + 255) / 256, 256, 0, stream>>>(W[l], wt[l], UNITS, UNITS, UNITS);
  }
  for (int l = 0; l < 4; ++l) {
    int n = UNITS * UNITS;
    transpose_convert_kernel<<<(n + 255) / 256, 256, 0, stream>>>(U[l], ut[l], UNITS, UNITS, UNITS);
  }

  // --- Embedding gather (time-major, f16, zero-padded) ---
  {
    size_t n = (size_t)SEQ * BATCH * EMB_PAD;
    embed_kernel<<<(unsigned)((n + 255) / 256), 256, 0, stream>>>(tokens, emb, xbuf);
  }

  // --- Zero initial hidden states (both ping-pong buffers for determinism) ---
  for (int l = 0; l < 4; ++l)
    for (int pp = 0; pp < 2; ++pp) {
      int n = BATCH * UNITS;
      zero_f16_kernel<<<(n + 255) / 256, 256, 0, stream>>>(h[l][pp], n);
    }

  // --- Sequential scan: 80 steps x 4 stacked layers, ping-pong h buffers ---
  dim3 sgrid(BATCH / 128, UNITS / 64); // (16, 8)
  for (int t = 0; t < SEQ; ++t) {
    const int p = t & 1;
    const _Float16* x_t = xbuf + (size_t)t * BATCH * EMB_PAD;
    rnn_step_kernel<<<sgrid, 256, 0, stream>>>(x_t, EMB_PAD, wt[0], h[0][p], ut[0], bv[0], h[0][p ^ 1]);
    rnn_step_kernel<<<sgrid, 256, 0, stream>>>(h[0][p ^ 1], UNITS, wt[1], h[1][p], ut[1], bv[1], h[1][p ^ 1]);
    rnn_step_kernel<<<sgrid, 256, 0, stream>>>(h[1][p ^ 1], UNITS, wt[2], h[2][p], ut[2], bv[2], h[2][p ^ 1]);
    rnn_step_kernel<<<sgrid, 256, 0, stream>>>(h[2][p ^ 1], UNITS, wt[3], h[3][p], ut[3], bv[3], h[3][p ^ 1]);
  }

  // --- Head: sigmoid(h4 @ Wo + bo) ---
  const _Float16* hfin = h[3][SEQ & 1]; // after t=79 (p=1), result lives in buffer 0
  output_kernel<<<(BATCH + 255) / 256, 256, 0, stream>>>(hfin, Wo, bo, out);
}
